// SelfAttention_84439057039753
// MI455X (gfx1250) — matrix operand order repaired and checked by reading, not compiled, not run on hardware
//
#include <hip/hip_runtime.h>
#include <hip/hip_bf16.h>

// ---------------------------------------------------------------------------
// Problem constants (B=2, S=2048, E=2048, H=16, D=128)
// ---------------------------------------------------------------------------
#define BATCH   2
#define SEQ     2048
#define EMB     2048
#define NHEAD   16
#define HDIM    128
#define MTOK    (BATCH * SEQ)          // 4096 rows
#define QKVN    (3 * EMB)              // 6144
#define RMS_EPS 1.1920928955078125e-07f
#define LOG2E   1.4426950408889634f

typedef __bf16 bf16;
typedef __attribute__((ext_vector_type(8)))  float v8f;
typedef __attribute__((ext_vector_type(16))) bf16  v16bf;

// 16x32 bf16 WMMA fragment = 8 VGPRs = two 16-byte pieces per lane.
union Frag {
    uint4 q[2];
    v16bf v;
};

__device__ __forceinline__ v8f wmma_bf16(const Frag& a, const Frag& b, v8f c) {
    return __builtin_amdgcn_wmma_f32_16x16x32_bf16(
        /*neg_a=*/false, a.v, /*neg_b=*/false, b.v,
        /*c_mod=*/(short)0, c, /*reuse_a=*/false, /*reuse_b=*/false);
}

// CDNA5 async global->LDS copy (16B per lane), tracked by ASYNCcnt.
__device__ __forceinline__ void async_copy_b128(unsigned lds_byte, const bf16* gptr) {
    asm volatile("global_load_async_to_lds_b128 %0, %1, off"
                 :: "v"(lds_byte), "v"(gptr)
                 : "memory");
}
#define S_WAIT_ASYNCCNT_0 asm volatile("s_wait_asynccnt 0x0" ::: "memory")

// ---------------------------------------------------------------------------
// Kernel 1: fp32 -> bf16 (vectorized x4)
// ---------------------------------------------------------------------------
__global__ __launch_bounds__(256) void f32_to_bf16_x4(
    const float* __restrict__ in, bf16* __restrict__ out, int n4)
{
    int i = blockIdx.x * blockDim.x + threadIdx.x;
    if (i >= n4) return;
    float4 f = reinterpret_cast<const float4*>(in)[i];
    union { bf16 h[4]; uint2 u; } o;
    o.h[0] = (bf16)f.x; o.h[1] = (bf16)f.y;
    o.h[2] = (bf16)f.z; o.h[3] = (bf16)f.w;
    reinterpret_cast<uint2*>(out)[i] = o.u;
}

// ---------------------------------------------------------------------------
// Kernel 2: GEMM  C[M,N] = A[M,K] * W[N,K]^T   (bf16 inputs, fp32 accum)
// 8 waves per block share ONE 64-column N-tile: the 64x32 bf16 weight panel
// for each 32-k stage is staged into LDS with global_load_async_to_lds_b128
// (one b128 per thread = whole 4KB panel), double-buffered on ASYNCcnt, so
// the async copy of stage s+1 overlaps the 16 WMMAs of stage s.  Each wave
// owns a 64-row M-subtile (4x4 accumulator array); A fragments use a 2-deep
// register pipeline of global_load_b128.
// ---------------------------------------------------------------------------
template <typename OutT>
__global__ __launch_bounds__(256) void gemm_bf16_wmma(
    const bf16* __restrict__ A, const bf16* __restrict__ W,
    OutT* __restrict__ C, int M, int N, int K)
{
    __shared__ bf16 bpanel[2][64][32];          // [stage][row][k]  (8 KB)

    const int tid  = threadIdx.x;
    const int lane = tid & 31;
    const int wave = tid >> 5;
    const int l16  = lane & 15;
    const int hf   = lane >> 4;

    const int tilesN = N >> 6;
    const int tn = blockIdx.x % tilesN;                  // shared by the block
    const int tm = (blockIdx.x / tilesN) * 8 + wave;     // per-wave M tile

    // Cooperative async B-panel copy: thread t moves 16B (row t/4, 8 elems).
    const int prow = tid >> 2;
    const int pcol = (tid & 3) * 8;
    const bf16* wsrc = W + (size_t)(tn * 64 + prow) * K + pcol;

    v8f acc[4][4];
#pragma unroll
    for (int i = 0; i < 4; ++i)
#pragma unroll
        for (int j = 0; j < 4; ++j) acc[i][j] = (v8f)0.0f;

    const bf16* arow[4];
#pragma unroll
    for (int i = 0; i < 4; ++i)
        arow[i] = A + (size_t)(tm * 64 + i * 16 + l16) * K;

    Frag a0[4], a1[4];

#define LOAD_A(AF, KOFF)                                                        \
    {                                                                           \
        _Pragma("unroll")                                                       \
        for (int i = 0; i < 4; ++i) {                                           \
            (AF)[i].q[0] = *reinterpret_cast<const uint4*>(arow[i] + (KOFF) + hf * 8);      \
            (AF)[i].q[1] = *reinterpret_cast<const uint4*>(arow[i] + (KOFF) + 16 + hf * 8); \
        }                                                                       \
    }

#define READ_B(BF, P)                                                           \
    {                                                                           \
        _Pragma("unroll")                                                       \
        for (int ni = 0; ni < 4; ++ni) {                                        \
            (BF)[ni].q[0] = *reinterpret_cast<const uint4*>(&bpanel[(P)][ni * 16 + l16][hf * 8]);      \
            (BF)[ni].q[1] = *reinterpret_cast<const uint4*>(&bpanel[(P)][ni * 16 + l16][16 + hf * 8]); \
        }                                                                       \
    }

#define WMMA_STAGE(AF, BF)                                                      \
    {                                                                           \
        _Pragma("unroll")                                                       \
        for (int mi = 0; mi < 4; ++mi)                                          \
            _Pragma("unroll")                                                   \
            for (int ni = 0; ni < 4; ++ni)                                      \
                acc[mi][ni] = wmma_bf16((AF)[mi], (BF)[ni], acc[mi][ni]);       \
    }

    // prologue: stage 0 async copy + first A fragments
    async_copy_b128((unsigned)(uintptr_t)&bpanel[0][prow][pcol], wsrc);
    LOAD_A(a0, 0)

    for (int k0 = 0; k0 < K; k0 += 64) {
        // ---- half-stage 0: consume buf0 (k0), copy k0+32 into buf1 ----
        S_WAIT_ASYNCCNT_0;
        __syncthreads();          // buf0 ready; readers of buf1 are done
        async_copy_b128((unsigned)(uintptr_t)&bpanel[1][prow][pcol], wsrc + k0 + 32);
        LOAD_A(a1, k0 + 32)
        {
            Frag b[4];
            READ_B(b, 0)
            WMMA_STAGE(a0, b)
        }
        // ---- half-stage 1: consume buf1 (k0+32), copy k0+64 into buf0 ----
        S_WAIT_ASYNCCNT_0;
        __syncthreads();
        if (k0 + 64 < K) {
            async_copy_b128((unsigned)(uintptr_t)&bpanel[0][prow][pcol], wsrc + k0 + 64);
            LOAD_A(a0, k0 + 64)
        }
        {
            Frag b[4];
            READ_B(b, 1)
            WMMA_STAGE(a1, b)
        }
    }
#undef LOAD_A
#undef READ_B
#undef WMMA_STAGE

    // C layout (ISA 7.12.2): vgpr j -> row j + 8*(lane>>4), col = lane&15.
#pragma unroll
    for (int mi = 0; mi < 4; ++mi)
#pragma unroll
        for (int ni = 0; ni < 4; ++ni) {
            int col = tn * 64 + ni * 16 + l16;
#pragma unroll
            for (int j = 0; j < 8; ++j) {
                int row = tm * 64 + mi * 16 + j + hf * 8;
                C[(size_t)row * N + col] = (OutT)acc[mi][ni][j];
            }
        }
}

// ---------------------------------------------------------------------------
// Kernel 3: split QKV, apply RoPE + RMS-norm to q,k (scale q by 1/sqrt(D)),
// write q,k as [B,H,S,D] bf16 and v transposed as [B,H,D,S] bf16.
// One wave per (token, head); each lane owns 4 elems = 2 rotary pairs.
// ---------------------------------------------------------------------------
__global__ __launch_bounds__(256) void rope_rms_split(
    const bf16* __restrict__ qkv, bf16* __restrict__ qb,
    bf16* __restrict__ kb, bf16* __restrict__ vT)
{
    const int wid  = blockIdx.x * 8 + (threadIdx.x >> 5);
    const int lane = threadIdx.x & 31;
    const int h = wid & (NHEAD - 1);
    const int m = wid >> 4;
    const int b = m >> 11;          // m / SEQ
    const int s = m & (SEQ - 1);

    const size_t base = (size_t)m * QKVN;
    const int d0 = lane * 4;        // elements d0..d0+3, pairs 2*lane, 2*lane+1

    // rate_p = 10000^(-p/128) = exp2(-p * (1/128) * log2(10000))
    const float kLog2_10k = 13.287712379549449f;
    float r0 = exp2f(-(float)(2 * lane)     * (1.0f / 128.0f) * kLog2_10k);
    float r1 = exp2f(-(float)(2 * lane + 1) * (1.0f / 128.0f) * kLog2_10k);
    float c0, s0, c1, s1;
    __sincosf((float)s * r0, &s0, &c0);
    __sincosf((float)s * r1, &s1, &c1);

    const size_t qkdst = ((size_t)(b * NHEAD + h) * SEQ + s) * HDIM + d0;

#pragma unroll
    for (int which = 0; which < 2; ++which) {   // 0 = q, 1 = k
        union { uint2 u; bf16 h4[4]; } iv;
        iv.u = *reinterpret_cast<const uint2*>(qkv + base + which * EMB + h * HDIM + d0);
        float x0 = (float)iv.h4[0], x1 = (float)iv.h4[1];
        float x2 = (float)iv.h4[2], x3 = (float)iv.h4[3];
        float lo0 = x0 * c0 - x1 * s0, ro0 = x0 * s0 + x1 * c0;
        float lo1 = x2 * c1 - x3 * s1, ro1 = x2 * s1 + x3 * c1;

        float ss = lo0 * lo0 + ro0 * ro0 + lo1 * lo1 + ro1 * ro1;
#pragma unroll
        for (int msk = 1; msk < 32; msk <<= 1) ss += __shfl_xor(ss, msk);
        float scl = rsqrtf(ss * (1.0f / 128.0f) + RMS_EPS);
        if (which == 0) scl *= 0.08838834764831845f;   // 1/sqrt(128) folded into q

        union { uint2 u; bf16 h4[4]; } ov;
        ov.h4[0] = (bf16)(lo0 * scl); ov.h4[1] = (bf16)(ro0 * scl);
        ov.h4[2] = (bf16)(lo1 * scl); ov.h4[3] = (bf16)(ro1 * scl);
        bf16* dst = (which == 0) ? qb : kb;
        *reinterpret_cast<uint2*>(dst + qkdst) = ov.u;
    }

    // v: plain copy, transposed to [B,H,D,S]
    union { uint2 u; bf16 h4[4]; } vv;
    vv.u = *reinterpret_cast<const uint2*>(qkv + base + 2 * EMB + h * HDIM + d0);
    const size_t vbase = (size_t)(b * NHEAD + h) * HDIM * SEQ;
#pragma unroll
    for (int t = 0; t < 4; ++t)
        vT[vbase + (size_t)(d0 + t) * SEQ + s] = vv.h4[t];
}

// ---------------------------------------------------------------------------
// Kernel 4: flash attention, one wave per (b, h, 32-query tile).
// scores: Q(32x128) x K^T via WMMA; online softmax in C-layout
// (row = j + 8*(lane>>4), key = lane&15 -> row reductions are shfl_xor 1/2/4/8
//  within each 16-lane group). P staged via LDS (bf16, stride 40 keeps
//  ds_load_b128 16B-aligned) to convert C-layout -> A-layout for P @ V.
// V fragments for the chunk are loaded BEFORE the softmax VALU section so the
// vmem latency overlaps the exp/shuffle work.
// ---------------------------------------------------------------------------
#define PSTRIDE 40
__global__ __launch_bounds__(32) void flash_attn(
    const bf16* __restrict__ qb, const bf16* __restrict__ kb,
    const bf16* __restrict__ vT, bf16* __restrict__ ob)
{
    __shared__ bf16 plds[32 * PSTRIDE];

    const int lane = threadIdx.x & 31;
    const int l16  = lane & 15;
    const int hf   = lane >> 4;

    const int id    = blockIdx.x;
    const int qt    = id & (SEQ / 32 - 1);          // 64 q-tiles
    const int h     = (id >> 6) & (NHEAD - 1);
    const int b     = id >> 10;
    const int qbase = qt * 32;

    const bf16* Q  = qb + (size_t)(b * NHEAD + h) * SEQ * HDIM;
    const bf16* Kp = kb + (size_t)(b * NHEAD + h) * SEQ * HDIM;
    const bf16* Vt = vT + (size_t)(b * NHEAD + h) * HDIM * SEQ;

    // Resident Q fragments: 2 row-subtiles x 4 K-steps.
    Frag qf[2][4];
#pragma unroll
    for (int mt = 0; mt < 2; ++mt) {
        const bf16* qrow = Q + (size_t)(qbase + mt * 16 + l16) * HDIM;
#pragma unroll
        for (int kt = 0; kt < 4; ++kt) {
            qf[mt][kt].q[0] = *reinterpret_cast<const uint4*>(qrow + kt * 32 + hf * 8);
            qf[mt][kt].q[1] = *reinterpret_cast<const uint4*>(qrow + kt * 32 + 16 + hf * 8);
        }
    }

    v8f oacc[2][8];
#pragma unroll
    for (int mt = 0; mt < 2; ++mt)
#pragma unroll
        for (int dt = 0; dt < 8; ++dt) oacc[mt][dt] = (v8f)0.0f;

    float mrun[2][8], lrun[2][8];
#pragma unroll
    for (int mt = 0; mt < 2; ++mt)
#pragma unroll
        for (int j = 0; j < 8; ++j) { mrun[mt][j] = -1e30f; lrun[mt][j] = 0.0f; }

    for (int kbase = 0; kbase <= qbase; kbase += 32) {
        // ---- scores S = Q K^T (2x2 tiles of 16x16) ----
        v8f sacc[2][2];
#pragma unroll
        for (int mt = 0; mt < 2; ++mt)
#pragma unroll
            for (int nt = 0; nt < 2; ++nt) sacc[mt][nt] = (v8f)0.0f;

#pragma unroll
        for (int kt = 0; kt < 4; ++kt) {
            Frag bk[2];
#pragma unroll
            for (int nt = 0; nt < 2; ++nt) {
                const bf16* krow = Kp + (size_t)(kbase + nt * 16 + l16) * HDIM;
                const uint4* p = reinterpret_cast<const uint4*>(krow + kt * 32 + hf * 8);
                bk[nt].q[0] = p[0];
                bk[nt].q[1] = p[2];
                if (kbase + 32 <= qbase)        // prefetch next chunk's K rows
                    __builtin_prefetch(krow + HDIM * 32 + kt * 32 + hf * 16, 0, 1);
            }
#pragma unroll
            for (int mt = 0; mt < 2; ++mt)
#pragma unroll
                for (int nt = 0; nt < 2; ++nt)
                    sacc[mt][nt] = wmma_bf16(qf[mt][kt], bk[nt], sacc[mt][nt]);
        }

        // ---- V fragments for this chunk: issue loads now, consume after
        //      softmax so vmem latency hides under the VALU work ----
        Frag vf[8];
#pragma unroll
        for (int dt = 0; dt < 8; ++dt) {
            const bf16* vrow = Vt + (size_t)(dt * 16 + l16) * SEQ + kbase;
            const uint4* p = reinterpret_cast<const uint4*>(vrow + hf * 8);
            vf[dt].q[0] = p[0];
            vf[dt].q[1] = p[2];
        }

        // ---- causal mask + online softmax update ----
#pragma unroll
        for (int mt = 0; mt < 2; ++mt) {
#pragma unroll
            for (int j = 0; j < 8; ++j) {
                int row  = qbase + mt * 16 + j + hf * 8;
                int key0 = kbase + l16;
                int key1 = kbase + 16 + l16;
                float v0 = (key0 <= row) ? sacc[mt][0][j] : -1e30f;
                float v1 = (key1 <= row) ? sacc[mt][1][j] : -1e30f;

                float mx = fmaxf(v0, v1);
                mx = fmaxf(mx, __shfl_xor(mx, 1));
                mx = fmaxf(mx, __shfl_xor(mx, 2));
                mx = fmaxf(mx, __shfl_xor(mx, 4));
                mx = fmaxf(mx, __shfl_xor(mx, 8));

                float mnew  = fmaxf(mrun[mt][j], mx);
                float alpha = exp2f((mrun[mt][j] - mnew) * LOG2E);
                mrun[mt][j] = mnew;

                float p0 = exp2f((v0 - mnew) * LOG2E);
                float p1 = exp2f((v1 - mnew) * LOG2E);
                float rs = p0 + p1;
                rs += __shfl_xor(rs, 1);
                rs += __shfl_xor(rs, 2);
                rs += __shfl_xor(rs, 4);
                rs += __shfl_xor(rs, 8);
                lrun[mt][j] = lrun[mt][j] * alpha + rs;

#pragma unroll
                for (int dt = 0; dt < 8; ++dt) oacc[mt][dt][j] *= alpha;

                int prow = mt * 16 + j + hf * 8;
                plds[prow * PSTRIDE + l16]      = (bf16)p0;
                plds[prow * PSTRIDE + 16 + l16] = (bf16)p1;
            }
        }
        __syncthreads();

        // ---- O += P @ V (P from LDS in A-layout, V already in registers) ----
        Frag pf[2];
#pragma unroll
        for (int mt = 0; mt < 2; ++mt) {
            int prow = mt * 16 + l16;
            pf[mt].q[0] = *reinterpret_cast<const uint4*>(&plds[prow * PSTRIDE + hf * 8]);
            pf[mt].q[1] = *reinterpret_cast<const uint4*>(&plds[prow * PSTRIDE + 16 + hf * 8]);
        }
#pragma unroll
        for (int dt = 0; dt < 8; ++dt)
#pragma unroll
            for (int mt = 0; mt < 2; ++mt)
                oacc[mt][dt] = wmma_bf16(pf[mt], vf[dt], oacc[mt][dt]);
    }

    // ---- finalize: divide by row sums, write [B,S,E] bf16 ----
#pragma unroll
    for (int mt = 0; mt < 2; ++mt)
#pragma unroll
        for (int dt = 0; dt < 8; ++dt) {
            int col = h * HDIM + dt * 16 + l16;
#pragma unroll
            for (int j = 0; j < 8; ++j) {
                int srow = qbase + mt * 16 + j + hf * 8;
                float o = oacc[mt][dt][j] / lrun[mt][j];
                ob[((size_t)(b * SEQ + srow)) * EMB + col] = (bf16)o;
            }
        }
}

// ---------------------------------------------------------------------------
// Launcher.  Workspace layout (bytes, all 256B-aligned); total ~168 MB:
//   xb     @ 0          16.8 MB   bf16 x            [4096,2048]
//   wqkvb  @ 16777216   25.2 MB   bf16 w_qkv        [6144,2048]
//   wprojb @ 41943040    8.4 MB   bf16 w_proj       [2048,2048]
//   qkvb   @ 50331648   50.3 MB   bf16 qkv          [4096,6144]
//   qbuf   @ 100663296  16.8 MB   bf16 q  [B,H,S,D]
//   kbuf   @ 117440512  16.8 MB   bf16 k  [B,H,S,D]
//   vT     @ 134217728  16.8 MB   bf16 v^T [B,H,D,S]
//   obuf   @ 150994944  16.8 MB   bf16 attn out [B,S,E]
// ---------------------------------------------------------------------------
extern "C" void kernel_launch(void* const* d_in, const int* in_sizes, int n_in,
                              void* d_out, int out_size, void* d_ws, size_t ws_size,
                              hipStream_t stream)
{
    const float* x      = (const float*)d_in[0];
    const float* w_qkv  = (const float*)d_in[1];
    const float* w_proj = (const float*)d_in[2];
    float* out = (float*)d_out;

    char* ws = (char*)d_ws;
    bf16* xb     = (bf16*)(ws + 0);
    bf16* wqkvb  = (bf16*)(ws + 16777216);
    bf16* wprojb = (bf16*)(ws + 41943040);
    bf16* qkvb   = (bf16*)(ws + 50331648);
    bf16* qbuf   = (bf16*)(ws + 100663296);
    bf16* kbuf   = (bf16*)(ws + 117440512);
    bf16* vT     = (bf16*)(ws + 134217728);
    bf16* obuf   = (bf16*)(ws + 150994944);

    // 1) fp32 -> bf16 converts
    {
        int n4 = MTOK * EMB / 4;                                  // x
        f32_to_bf16_x4<<<(n4 + 255) / 256, 256, 0, stream>>>(x, xb, n4);
        n4 = QKVN * EMB / 4;                                      // w_qkv
        f32_to_bf16_x4<<<(n4 + 255) / 256, 256, 0, stream>>>(w_qkv, wqkvb, n4);
        n4 = EMB * EMB / 4;                                       // w_proj
        f32_to_bf16_x4<<<(n4 + 255) / 256, 256, 0, stream>>>(w_proj, wprojb, n4);
    }

    // 2) qkv = x @ w_qkv^T   (4096 x 6144 x 2048) -> bf16
    {
        int tiles = (MTOK / 64) * (QKVN / 64);                    // 6144
        gemm_bf16_wmma<bf16><<<tiles / 8, 256, 0, stream>>>(
            xb, wqkvb, qkvb, MTOK, QKVN, EMB);
    }

    // 3) RoPE + RMS-norm + split/transpose
    rope_rms_split<<<(MTOK * NHEAD) / 8, 256, 0, stream>>>(qkvb, qbuf, kbuf, vT);

    // 4) causal flash attention
    flash_attn<<<BATCH * NHEAD * (SEQ / 32), 32, 0, stream>>>(qbuf, kbuf, vT, obuf);

    // 5) out = attn @ w_proj^T  (4096 x 2048 x 2048) -> fp32 d_out
    {
        int tiles = (MTOK / 64) * (EMB / 64);                     // 2048
        gemm_bf16_wmma<float><<<tiles / 8, 256, 0, stream>>>(
            obuf, wprojb, out, MTOK, EMB, EMB);
    }
}
